// SlidingWindowCausalSelfAttention_80556406603855
// MI455X (gfx1250) — matrix operand order repaired and checked by reading, not compiled, not run on hardware
//
#include <hip/hip_runtime.h>
#include <hip/hip_bf16.h>

#define DIMN   1024
#define NHEADS 16
#define HDIM   64
#define WIN    512
#define BATCH  2
#define SEQ    2048

typedef __attribute__((ext_vector_type(16))) _Float16 v16h;
typedef __attribute__((ext_vector_type(8)))  _Float16 v8h;
typedef __attribute__((ext_vector_type(8)))  float    v8f;
typedef __attribute__((ext_vector_type(4)))  float    v4f;

static __device__ __forceinline__ v8f wmma16(v16h a, v16h b, v8f c) {
  return __builtin_amdgcn_wmma_f32_16x16x32_f16(false, a, false, b, (short)0, c, false, false);
}
static __device__ __forceinline__ float redmax16(float v) {
#pragma unroll
  for (int m = 1; m < 16; m <<= 1) v = fmaxf(v, __shfl_xor(v, m, 16));
  return v;
}
static __device__ __forceinline__ float redsum16(float v) {
#pragma unroll
  for (int m = 1; m < 16; m <<= 1) v += __shfl_xor(v, m, 16);
  return v;
}
// WMMA A-operand (16xK f16 tile, K-chunk of 32): row = lane%16,
// elements 0..15 -> K = k0 + g*16 + e (contiguous per half-wave)
static __device__ __forceinline__ v16h load_a16(const _Float16* row, int k0, int g) {
  v8h lo = *(const v8h*)(row + k0 + g * 16);
  v8h hi = *(const v8h*)(row + k0 + g * 16 + 8);
  v16h a;
#pragma unroll
  for (int e = 0; e < 8; ++e) { a[e] = lo[e]; a[8 + e] = hi[e]; }
  return a;
}

// ---------------- one-shot f32 -> f16 conversion ----------------
__global__ __launch_bounds__(256)
void cvt_kernel(const float* __restrict__ src, _Float16* __restrict__ dst, int n) {
  const int i = (blockIdx.x * 256 + threadIdx.x) * 8;
  if (i < n) {
    v4f u0 = *(const v4f*)(src + i);
    v4f u1 = *(const v4f*)(src + i + 4);
    v8h o;
#pragma unroll
    for (int e = 0; e < 4; ++e) { o[e] = (_Float16)u0[e]; o[4 + e] = (_Float16)u1[e]; }
    *(v8h*)(dst + i) = o;
  }
}

// ---------------- QKV projection + RMSNorm + RoPE + gain ----------------
// grid: (NTOK/32, NHEADS, 3{q,k,v}); block: 32 (one wave)
// M=32 per wave: two 16-row blocks share every B (weight) tile.
__global__ __launch_bounds__(32)
void qkv_kernel(const _Float16* __restrict__ xh, const _Float16* __restrict__ Wqh,
                const _Float16* __restrict__ Wkh, const _Float16* __restrict__ Wvh,
                const float* __restrict__ qgain,
                _Float16* __restrict__ qb, _Float16* __restrict__ kb,
                _Float16* __restrict__ vt) {
  const int rt = blockIdx.x, h = blockIdx.y, mat = blockIdx.z;
  const int lane = threadIdx.x, g = lane >> 4, n16 = lane & 15;
  const _Float16* W = (mat == 0) ? Wqh : (mat == 1) ? Wkh : Wvh;
  const _Float16* xrow0 = xh + (size_t)(rt * 32 + n16) * DIMN;
  const _Float16* xrow1 = xrow0 + (size_t)16 * DIMN;
  const _Float16* wrow = W + (size_t)(h * HDIM + n16) * DIMN;

  v8f c[2][4] = {};
  for (int k0 = 0; k0 < DIMN; k0 += 32) {
    const v16h a0 = load_a16(xrow0, k0, g);
    const v16h a1 = load_a16(xrow1, k0, g);
#pragma unroll
    for (int dt = 0; dt < 4; ++dt) {
      // B: col n = lane%16 -> W row h*64 + dt*16 + n16, K contiguous per half-wave
      const v16h b = *(const v16h*)(wrow + (size_t)dt * 16 * DIMN + k0 + g * 16);
      c[0][dt] = wmma16(a0, b, c[0][dt]);
      c[1][dt] = wmma16(a1, b, c[1][dt]);
    }
  }

#pragma unroll
  for (int rb = 0; rb < 2; ++rb) {
    const int rowbase = rt * 32 + rb * 16;
    if (mat < 2) {
      // RMS norm over the 64-dim head (row lives across 16 lanes x 4 tiles)
#pragma unroll
      for (int r = 0; r < 8; ++r) {
        float ss = 0.f;
#pragma unroll
        for (int dt = 0; dt < 4; ++dt) ss += c[rb][dt][r] * c[rb][dt][r];
        ss = redsum16(ss);
        const float inv = rsqrtf(ss * (1.0f / HDIM) + 1.1920929e-7f);
#pragma unroll
        for (int dt = 0; dt < 4; ++dt) c[rb][dt][r] *= inv;
      }
      // RoPE: pair (d, d+32) lives in tiles (t, t+2), same lane/row
#pragma unroll
      for (int t = 0; t < 2; ++t) {
        const float f = exp2f(-(float)(t * 16 + n16) * 0.4152410118609203f); // log2(1e4)/32
#pragma unroll
        for (int r = 0; r < 8; ++r) {
          const int tok = rowbase + g * 8 + r;
          const float ang = (float)(tok & (SEQ - 1)) * f;
          float sn, cs;
          sincosf(ang, &sn, &cs);
          const float t1 = c[rb][t][r], t2 = c[rb][t + 2][r];
          c[rb][t][r]     =  t1 * cs + t2 * sn;
          c[rb][t + 2][r] = -t1 * sn + t2 * cs;
        }
      }
      const float gn = (mat == 0) ? qgain[h] : 1.0f;
      _Float16* dst = (mat == 0) ? qb : kb;
#pragma unroll
      for (int dt = 0; dt < 4; ++dt) {
        const int d = dt * 16 + n16;
#pragma unroll
        for (int r = 0; r < 8; ++r) {
          const int tok = rowbase + g * 8 + r;
          const int bb = tok >> 11, s = tok & (SEQ - 1);
          dst[(((size_t)bb * NHEADS + h) * SEQ + s) * HDIM + d] = (_Float16)(c[rb][dt][r] * gn);
        }
      }
    } else {
      // V stored transposed: vt[b][h][d][s] so PV B-operand is contiguous
      const int tok0 = rowbase + g * 8;
      const int bb = tok0 >> 11, s0 = tok0 & (SEQ - 1);
#pragma unroll
      for (int dt = 0; dt < 4; ++dt) {
        const int d = dt * 16 + n16;
        _Float16* p = vt + (((size_t)bb * NHEADS + h) * HDIM + d) * SEQ + s0;
        v8h tmp;
#pragma unroll
        for (int r = 0; r < 8; ++r) tmp[r] = (_Float16)c[rb][dt][r];
        *(v8h*)p = tmp;
      }
    }
  }
}

// ---------------- Sliding-window flash attention ----------------
// grid: (SEQ/16, NHEADS, BATCH); block: 32 (one wave)
__global__ __launch_bounds__(32)
void attn_kernel(const _Float16* __restrict__ qb, const _Float16* __restrict__ kb,
                 const _Float16* __restrict__ vt, _Float16* __restrict__ att) {
  __shared__ _Float16 lds_p[16 * 32];
  const int qt = blockIdx.x, h = blockIdx.y, b = blockIdx.z;
  const int lane = threadIdx.x, g = lane >> 4, n16 = lane & 15;
  const int qbase = qt * 16;
  const _Float16* qm = qb + ((size_t)b * NHEADS + h) * SEQ * HDIM;
  const _Float16* km = kb + ((size_t)b * NHEADS + h) * SEQ * HDIM;
  const _Float16* vm = vt + ((size_t)b * NHEADS + h) * HDIM * SEQ;

  v16h aq[2];
  {
    const _Float16* qr = qm + (size_t)(qbase + n16) * HDIM;
#pragma unroll
    for (int cc = 0; cc < 2; ++cc) aq[cc] = load_a16(qr, cc * 32, g);
  }

  v8f o[4] = {};
  float mr[8], lr[8];
#pragma unroll
  for (int r = 0; r < 8; ++r) { mr[r] = -__builtin_inff(); lr[r] = 0.f; }

  const int lowk = qbase - (WIN - 1);
  const int ktlo = lowk > 0 ? (lowk >> 4) : 0;
  for (int kt = ktlo; kt <= qt; kt += 2) {
    const int kb16 = kt * 16;
    const bool two = (kt + 1) <= qt;

    v8f s0 = {}, s1 = {};
    {
      const _Float16* kr0 = km + (size_t)(kb16 + n16) * HDIM;
#pragma unroll
      for (int cc = 0; cc < 2; ++cc)
        s0 = wmma16(aq[cc], *(const v16h*)(kr0 + cc * 32 + g * 16), s0);
    }
    if (two) {
      const _Float16* kr1 = km + (size_t)(kb16 + 16 + n16) * HDIM;
#pragma unroll
      for (int cc = 0; cc < 2; ++cc)
        s1 = wmma16(aq[cc], *(const v16h*)(kr1 + cc * 32 + g * 16), s1);
    }

    const int col0 = kb16 + n16, col1 = kb16 + 16 + n16;
    float p0[8], p1[8];
#pragma unroll
    for (int r = 0; r < 8; ++r) {
      const int row = qbase + g * 8 + r;
      const float v0 = (col0 <= row && col0 >= row - (WIN - 1)) ? s0[r] * 0.125f : -1e30f;
      const float v1 = (two && col1 <= row && col1 >= row - (WIN - 1)) ? s1[r] * 0.125f : -1e30f;
      const float mc = redmax16(fmaxf(v0, v1));
      const float mn = fmaxf(mr[r], mc);
      const float alpha = expf(mr[r] - mn);
      p0[r] = expf(v0 - mn);
      p1[r] = expf(v1 - mn);
      const float rs = redsum16(p0[r] + p1[r]);
      lr[r] = lr[r] * alpha + rs;
      mr[r] = mn;
#pragma unroll
      for (int dt = 0; dt < 4; ++dt) o[dt][r] *= alpha;
    }

    // C-layout -> A-layout transpose of probs through LDS
    __syncthreads();
#pragma unroll
    for (int r = 0; r < 8; ++r) {
      lds_p[(g * 8 + r) * 32 + n16]      = (_Float16)p0[r];
      lds_p[(g * 8 + r) * 32 + 16 + n16] = (_Float16)p1[r];
    }
    __syncthreads();
    v16h ap;
#pragma unroll
    for (int e = 0; e < 8; ++e) {
      ap[e]     = lds_p[n16 * 32 + g * 16 + e];
      ap[8 + e] = lds_p[n16 * 32 + g * 16 + 8 + e];
    }

#pragma unroll
    for (int dt = 0; dt < 4; ++dt) {
      const int d = dt * 16 + n16;
      v16h bv = {};
      if (two || g == 0)  // guard OOB tail; probs are 0 for guarded keys
        bv = *(const v16h*)(vm + (size_t)d * SEQ + kb16 + g * 16);
      o[dt] = wmma16(ap, bv, o[dt]);
    }
  }

#pragma unroll
  for (int dt = 0; dt < 4; ++dt) {
    const int d = h * HDIM + dt * 16 + n16;
#pragma unroll
    for (int r = 0; r < 8; ++r) {
      const int s = qbase + g * 8 + r;
      att[((size_t)b * SEQ + s) * DIMN + d] = (_Float16)(o[dt][r] / lr[r]);
    }
  }
}

// ---------------- Output projection ----------------
// grid: (NTOK/32, DIMN/64); block: 32 (one wave); M=32 per wave
__global__ __launch_bounds__(32)
void proj_kernel(const _Float16* __restrict__ att, const _Float16* __restrict__ Wph,
                 float* __restrict__ out) {
  const int rt = blockIdx.x, cg = blockIdx.y;
  const int lane = threadIdx.x, g = lane >> 4, n16 = lane & 15;
  const _Float16* yr0 = att + (size_t)(rt * 32 + n16) * DIMN;
  const _Float16* yr1 = yr0 + (size_t)16 * DIMN;
  const _Float16* wrow = Wph + (size_t)(cg * 64 + n16) * DIMN;

  v8f c[2][4] = {};
  for (int k0 = 0; k0 < DIMN; k0 += 32) {
    const v16h a0 = load_a16(yr0, k0, g);
    const v16h a1 = load_a16(yr1, k0, g);
#pragma unroll
    for (int dt = 0; dt < 4; ++dt) {
      const v16h b = *(const v16h*)(wrow + (size_t)dt * 16 * DIMN + k0 + g * 16);
      c[0][dt] = wmma16(a0, b, c[0][dt]);
      c[1][dt] = wmma16(a1, b, c[1][dt]);
    }
  }
#pragma unroll
  for (int rb = 0; rb < 2; ++rb)
#pragma unroll
    for (int dt = 0; dt < 4; ++dt)
#pragma unroll
      for (int r = 0; r < 8; ++r)
        out[(size_t)(rt * 32 + rb * 16 + g * 8 + r) * DIMN + cg * 64 + dt * 16 + n16] =
            c[rb][dt][r];
}

extern "C" void kernel_launch(void* const* d_in, const int* in_sizes, int n_in,
                              void* d_out, int out_size, void* d_ws, size_t ws_size,
                              hipStream_t stream) {
  const float* x  = (const float*)d_in[0];
  const float* Wq = (const float*)d_in[1];
  const float* Wk = (const float*)d_in[2];
  const float* Wv = (const float*)d_in[3];
  const float* Wp = (const float*)d_in[4];
  const float* qg = (const float*)d_in[5];
  float* out = (float*)d_out;

  const size_t nx = (size_t)BATCH * SEQ * DIMN;             // 4,194,304
  const size_t nw = (size_t)DIMN * DIMN;                    // 1,048,576
  const size_t per = (size_t)BATCH * NHEADS * SEQ * HDIM;   // == nx

  _Float16* xh   = (_Float16*)d_ws;
  _Float16* wq16 = xh + nx;
  _Float16* wk16 = wq16 + nw;
  _Float16* wv16 = wk16 + nw;
  _Float16* wp16 = wv16 + nw;
  _Float16* qb   = wp16 + nw;
  _Float16* kb   = qb + per;
  _Float16* vt   = kb + per;
  _Float16* att  = vt + per;

  // one-shot f32 -> f16 materialization (activations + weights)
  cvt_kernel<<<dim3((int)(nx / 8 / 256)), 256, 0, stream>>>(x, xh, (int)nx);
  cvt_kernel<<<dim3((int)(nw / 8 / 256)), 256, 0, stream>>>(Wq, wq16, (int)nw);
  cvt_kernel<<<dim3((int)(nw / 8 / 256)), 256, 0, stream>>>(Wk, wk16, (int)nw);
  cvt_kernel<<<dim3((int)(nw / 8 / 256)), 256, 0, stream>>>(Wv, wv16, (int)nw);
  cvt_kernel<<<dim3((int)(nw / 8 / 256)), 256, 0, stream>>>(Wp, wp16, (int)nw);

  const int ntok_tiles32 = (BATCH * SEQ) / 32; // 128
  qkv_kernel<<<dim3(ntok_tiles32, NHEADS, 3), 32, 0, stream>>>(xh, wq16, wk16, wv16, qg, qb, kb, vt);
  attn_kernel<<<dim3(SEQ / 16, NHEADS, BATCH), 32, 0, stream>>>(qb, kb, vt, att);
  proj_kernel<<<dim3(ntok_tiles32, DIMN / 64), 32, 0, stream>>>(att, wp16, out);
}
